// GraphSAGE_19920058319553
// MI455X (gfx1250) — matrix operand order repaired and checked by reading, not compiled, not run on hardware
//
#include <hip/hip_runtime.h>
#include <hip/hip_bf16.h>
#include <stdint.h>

#define DH 128  // hidden width / input width (K dimension is always 128)

typedef __attribute__((ext_vector_type(16))) __bf16 v16bf;
typedef __attribute__((ext_vector_type(8)))  float  v8f;

__device__ __forceinline__ unsigned short f32_to_bf16(float f) {
    unsigned int x = __float_as_uint(f);
    unsigned int r = x + 0x7FFFu + ((x >> 16) & 1u);   // round-to-nearest-even
    return (unsigned short)(r >> 16);
}
__device__ __forceinline__ float bf16_to_f32(unsigned short u) {
    return __uint_as_float(((unsigned int)u) << 16);
}

union BF16Vec { v16bf v; unsigned short u[16]; uint4 q[2]; };
union U4x8    { uint4  q; unsigned short u[8];  };
union F4x8    { float4 f4[2]; float f[8];  };

// ---------------------------------------------------------------- utilities
__global__ void zero_f32_kernel(float* __restrict__ p, int n) {
    int i = blockIdx.x * blockDim.x + threadIdx.x;
    if (i < n) p[i] = 0.0f;
}

__global__ void deg_kernel(const int* __restrict__ dst, float* __restrict__ deg, int ne) {
    int i = blockIdx.x * blockDim.x + threadIdx.x;
    if (i < ne) atomicAdd(&deg[dst[i]], 1.0f);
}

__global__ void invdeg_kernel(float* __restrict__ deg, int n) {
    int i = blockIdx.x * blockDim.x + threadIdx.x;
    if (i < n) deg[i] = 1.0f / fmaxf(deg[i], 1.0f);
}

// f32 -> bf16, 4 elements per thread (used for features AND weight matrices)
__global__ void cvt_f32_bf16_kernel(const float4* __restrict__ in, uint2* __restrict__ out, int n4) {
    int i = blockIdx.x * blockDim.x + threadIdx.x;
    if (i >= n4) return;
    float4 f = in[i];
    uint2 o;
    o.x = (unsigned int)f32_to_bf16(f.x) | ((unsigned int)f32_to_bf16(f.y) << 16);
    o.y = (unsigned int)f32_to_bf16(f.z) | ((unsigned int)f32_to_bf16(f.w) << 16);
    out[i] = o;
}

// ------------------------------------------------- edge gather / scatter-add
// one thread handles 8 features of one edge: bf16 gather (16B) + 8 f32 atomics.
// node table (25.6 MB bf16) and agg (51.2 MB f32) are both L2-resident.
__global__ void scatter_kernel(const unsigned short* __restrict__ hbf,
                               const int* __restrict__ src,
                               const int* __restrict__ dst,
                               float* __restrict__ agg, int ne) {
    int t = blockIdx.x * blockDim.x + threadIdx.x;
    int e = t >> 4;           // DH/8 = 16 chunks per edge
    int c = t & 15;
    if (e >= ne) return;
    int s = src[e], d = dst[e];
    U4x8 chunk;
    chunk.q = *(const uint4*)(hbf + (size_t)s * DH + c * 8);
    float* ap = agg + (size_t)d * DH + c * 8;
#pragma unroll
    for (int i = 0; i < 8; ++i)
        atomicAdd(ap + i, bf16_to_f32(chunk.u[i]));
}

// ------------------------------------------------------------- fused GEMM
// out[m,n] = sum_k h[m,k]*Wself[n,k] + bself[n]
//          + sum_k (agg[m,k]*invdeg[m])*Wneigh[n,k] + bneigh[n]   (+ReLU)
// one wave per 16-row node tile; K = 128 = 4 wmma k-steps; weights pre-bf16.
template <int NOUT, bool RELU, bool WRITE_BF>
__global__ void sage_gemm_kernel(const unsigned short* __restrict__ hbf,
                                 const float* __restrict__ agg,
                                 const float* __restrict__ invdeg,
                                 const unsigned short* __restrict__ Wself,   // bf16 [NOUT, DH]
                                 const float* __restrict__ bself,
                                 const unsigned short* __restrict__ Wneigh,  // bf16 [NOUT, DH]
                                 const float* __restrict__ bneigh,
                                 float* __restrict__ outf,
                                 unsigned short* __restrict__ outbf,
                                 int nnodes) {
    const int lane = threadIdx.x & 31;
    const int wave = threadIdx.x >> 5;
    const int tile = blockIdx.x * (blockDim.x >> 5) + wave;
    const int m0 = tile * 16;
    if (m0 >= nnodes) return;            // wave-uniform: EXEC stays all-ones

    const int mrow = m0 + (lane & 15);
    const int kh = lane >> 4;            // which K-half this lane supplies
    const float idg = invdeg[mrow];

    // hoist A operands (self: bf16 direct; neigh: f32 agg * invdeg -> bf16)
    BF16Vec a_self[4], a_neigh[4];
#pragma unroll
    for (int s = 0; s < 4; ++s) {
        const size_t base = (size_t)mrow * DH + s * 32 + kh * 8;
        a_self[s].q[0] = *(const uint4*)(hbf + base);        // K = 32s + 8kh .. +7
        a_self[s].q[1] = *(const uint4*)(hbf + base + 16);   // K = 32s + 16 + 8kh .. +7
        F4x8 x0, x1;
        x0.f4[0] = *(const float4*)(agg + base);
        x0.f4[1] = *(const float4*)(agg + base + 4);
        x1.f4[0] = *(const float4*)(agg + base + 16);
        x1.f4[1] = *(const float4*)(agg + base + 20);
#pragma unroll
        for (int i = 0; i < 8; ++i) {
            a_neigh[s].u[i]     = f32_to_bf16(x0.f[i] * idg);
            a_neigh[s].u[8 + i] = f32_to_bf16(x1.f[i] * idg);
        }
    }

    const int NT = NOUT / 16;
    for (int c = 0; c < NT; ++c) {
        const int n = c * 16 + (lane & 15);        // output column for this lane
        v8f acc = {};
#pragma unroll
        for (int s = 0; s < 4; ++s) {              // self term: pure load + wmma
            BF16Vec b;                              // B: K = 32s + 8kh .. +7, 32s + 16 + 8kh .. +7, col n
            const unsigned short* wp = Wself + (size_t)n * DH + s * 32 + kh * 8;
            b.q[0] = *(const uint4*)(wp);
            b.q[1] = *(const uint4*)(wp + 16);
            acc = __builtin_amdgcn_wmma_f32_16x16x32_bf16(
                      false, a_self[s].v, false, b.v, (short)0, acc, false, false);
        }
#pragma unroll
        for (int s = 0; s < 4; ++s) {              // neighbor term
            BF16Vec b;
            const unsigned short* wp = Wneigh + (size_t)n * DH + s * 32 + kh * 8;
            b.q[0] = *(const uint4*)(wp);
            b.q[1] = *(const uint4*)(wp + 16);
            acc = __builtin_amdgcn_wmma_f32_16x16x32_bf16(
                      false, a_neigh[s].v, false, b.v, (short)0, acc, false, false);
        }
        const float bias = bself[n] + bneigh[n];
#pragma unroll
        for (int r = 0; r < 8; ++r) {
            const int m = m0 + kh * 8 + r;         // C/D layout: M = 8*(lane>=16)+r
            float val = acc[r] + bias;
            if (RELU) val = fmaxf(val, 0.0f);
            if (WRITE_BF) outbf[(size_t)m * NOUT + n] = f32_to_bf16(val);
            else          outf [(size_t)m * NOUT + n] = val;
        }
    }
}

// ------------------------------------------------------------------ driver
static inline int ceil_div(long long a, int b) { return (int)((a + b - 1) / b); }
static inline size_t align256(size_t x) { return (x + 255) & ~(size_t)255; }

extern "C" void kernel_launch(void* const* d_in, const int* in_sizes, int n_in,
                              void* d_out, int out_size, void* d_ws, size_t ws_size,
                              hipStream_t stream) {
    const float* features = (const float*)d_in[0];
    const int*   src      = (const int*)d_in[1];
    const int*   dst      = (const int*)d_in[2];
    const float* Ws0 = (const float*)d_in[3];  const float* bs0 = (const float*)d_in[4];
    const float* Wn0 = (const float*)d_in[5];  const float* bn0 = (const float*)d_in[6];
    const float* Ws1 = (const float*)d_in[7];  const float* bs1 = (const float*)d_in[8];
    const float* Wn1 = (const float*)d_in[9];  const float* bn1 = (const float*)d_in[10];
    const float* Ws2 = (const float*)d_in[11]; const float* bs2 = (const float*)d_in[12];
    const float* Wn2 = (const float*)d_in[13]; const float* bn2 = (const float*)d_in[14];

    const int nn = in_sizes[0] / DH;   // 100000 nodes
    const int ne = in_sizes[1];        // 1600000 edges
    const int NOUT = 64;

    // workspace carve-up
    char* ws = (char*)d_ws;
    size_t o = 0;
    float* agg = (float*)(ws + o);          o += align256((size_t)nn * DH * sizeof(float));
    float* deg = (float*)(ws + o);          o += align256((size_t)nn * sizeof(float));
    unsigned short* hA = (unsigned short*)(ws + o); o += align256((size_t)nn * DH * sizeof(unsigned short));
    unsigned short* hB = (unsigned short*)(ws + o); o += align256((size_t)nn * DH * sizeof(unsigned short));
    // bf16 weight copies (converted once, reused by all 6250 row tiles)
    unsigned short* Ws0b = (unsigned short*)(ws + o); o += align256((size_t)DH * DH * 2);
    unsigned short* Wn0b = (unsigned short*)(ws + o); o += align256((size_t)DH * DH * 2);
    unsigned short* Ws1b = (unsigned short*)(ws + o); o += align256((size_t)DH * DH * 2);
    unsigned short* Wn1b = (unsigned short*)(ws + o); o += align256((size_t)DH * DH * 2);
    unsigned short* Ws2b = (unsigned short*)(ws + o); o += align256((size_t)NOUT * DH * 2);
    unsigned short* Wn2b = (unsigned short*)(ws + o); o += align256((size_t)NOUT * DH * 2);
    (void)ws_size;

    const int T = 256;
    const int nfeat  = nn * DH;
    const int tiles  = ceil_div(nn, 16);
    const int gblk   = ceil_div(tiles, T / 32);   // 8 row-tiles (waves) per block

    // degree (topology-only; shared by all layers), then invert in place
    zero_f32_kernel<<<ceil_div(nn, T), T, 0, stream>>>(deg, nn);
    deg_kernel<<<ceil_div(ne, T), T, 0, stream>>>(dst, deg, ne);
    invdeg_kernel<<<ceil_div(nn, T), T, 0, stream>>>(deg, nn);

    // features f32 -> bf16 (L2-resident gather table, also WMMA A operand)
    cvt_f32_bf16_kernel<<<ceil_div(nfeat / 4, T), T, 0, stream>>>(
        (const float4*)features, (uint2*)hA, nfeat / 4);

    // weights f32 -> bf16 (once)
    {
        const int wq = DH * DH / 4;          // 4096 quads
        const int wq2 = NOUT * DH / 4;       // 2048 quads
        cvt_f32_bf16_kernel<<<ceil_div(wq,  T), T, 0, stream>>>((const float4*)Ws0, (uint2*)Ws0b, wq);
        cvt_f32_bf16_kernel<<<ceil_div(wq,  T), T, 0, stream>>>((const float4*)Wn0, (uint2*)Wn0b, wq);
        cvt_f32_bf16_kernel<<<ceil_div(wq,  T), T, 0, stream>>>((const float4*)Ws1, (uint2*)Ws1b, wq);
        cvt_f32_bf16_kernel<<<ceil_div(wq,  T), T, 0, stream>>>((const float4*)Wn1, (uint2*)Wn1b, wq);
        cvt_f32_bf16_kernel<<<ceil_div(wq2, T), T, 0, stream>>>((const float4*)Ws2, (uint2*)Ws2b, wq2);
        cvt_f32_bf16_kernel<<<ceil_div(wq2, T), T, 0, stream>>>((const float4*)Wn2, (uint2*)Wn2b, wq2);
    }

    // ---- layer 0: hA -> hB
    zero_f32_kernel<<<ceil_div(nfeat, T), T, 0, stream>>>(agg, nfeat);
    scatter_kernel<<<ceil_div((long long)ne * 16, T), T, 0, stream>>>(hA, src, dst, agg, ne);
    sage_gemm_kernel<DH, true, true><<<gblk, T, 0, stream>>>(
        hA, agg, deg, Ws0b, bs0, Wn0b, bn0, nullptr, hB, nn);

    // ---- layer 1: hB -> hA
    zero_f32_kernel<<<ceil_div(nfeat, T), T, 0, stream>>>(agg, nfeat);
    scatter_kernel<<<ceil_div((long long)ne * 16, T), T, 0, stream>>>(hB, src, dst, agg, ne);
    sage_gemm_kernel<DH, true, true><<<gblk, T, 0, stream>>>(
        hB, agg, deg, Ws1b, bs1, Wn1b, bn1, nullptr, hA, nn);

    // ---- layer 2: hA -> d_out (f32, no relu)
    zero_f32_kernel<<<ceil_div(nfeat, T), T, 0, stream>>>(agg, nfeat);
    scatter_kernel<<<ceil_div((long long)ne * 16, T), T, 0, stream>>>(hA, src, dst, agg, ne);
    sage_gemm_kernel<NOUT, false, false><<<gblk, T, 0, stream>>>(
        hA, agg, deg, Ws2b, bs2, Wn2b, bn2, (float*)d_out, nullptr, nn);
    (void)out_size; (void)n_in;
}
